// MVGCN_9070970929428
// MI455X (gfx1250) — matrix operand order repaired and checked by reading, not compiled, not run on hardware
//
#include <hip/hip_runtime.h>
#include <hip/hip_bf16.h>

#define NROW 10000
#define KP   10016      // padded K / row pitch (multiple of 32)
#define EDIM 64
#define BPAIR 16384
#define LAM1 0.001f
#define LAM2 1.0f

typedef __attribute__((ext_vector_type(16))) __bf16 v16bf;
typedef __attribute__((ext_vector_type(8)))  __bf16 v8bf;
typedef __attribute__((ext_vector_type(8)))  float  v8f;

__device__ __forceinline__ __bf16 to_bf16(float f) {
    unsigned u = __float_as_uint(f);
    u += 0x7FFFu + ((u >> 16) & 1u);          // round-to-nearest-even
    union { unsigned short s; __bf16 b; } cv;
    cv.s = (unsigned short)(u >> 16);
    return cv.b;
}
__device__ __forceinline__ float from_bf16(__bf16 b) {
    union { __bf16 b; unsigned short s; } cv; cv.b = b;
    return __uint_as_float(((unsigned)cv.s) << 16);
}

__device__ __forceinline__ float block_reduce_sum(float s) {
    for (int off = 16; off; off >>= 1) s += __shfl_down(s, off, 32);
    __shared__ float sm[8];
    int lane = threadIdx.x & 31, w = threadIdx.x >> 5;
    if (lane == 0) sm[w] = s;
    __syncthreads();
    if (w == 0) {
        s = (lane < 8) ? sm[lane] : 0.0f;
        for (int off = 4; off; off >>= 1) s += __shfl_down(s, off, 32);
    }
    return s;   // valid in thread 0
}

// ---------------------------------------------------------------------------
// Pass 0: row sums of A (-> dinv) fused with fp32 -> bf16 conversion, padded.
// ---------------------------------------------------------------------------
__global__ __launch_bounds__(256)
void rowsum_convert(const float* __restrict__ A, __bf16* __restrict__ Abf,
                    float* __restrict__ dinv) {
    int row = blockIdx.x;
    const float* src = A + (size_t)row * NROW;
    __bf16* dst = Abf + (size_t)row * KP;
    float s = 0.0f;
    for (int c = threadIdx.x; c < KP; c += 256) {
        float v = (c < NROW) ? src[c] : 0.0f;
        dst[c] = to_bf16(v);
        s += v;
    }
    s = block_reduce_sum(s);
    if (threadIdx.x == 0)
        dinv[row] = (s > 0.0f) ? rsqrtf(s) : 0.0f;
}

// fp32 [R x 64] row-major -> bf16 [64 x RP] transposed, zero-padded rows.
__global__ void convert_T(const float* __restrict__ src, __bf16* __restrict__ dst,
                          int R, int RP) {
    int total = RP * EDIM;
    for (int idx = blockIdx.x * blockDim.x + threadIdx.x; idx < total;
         idx += gridDim.x * blockDim.x) {
        int col = idx / RP, row = idx - col * RP;
        float v = (row < R) ? src[(size_t)row * EDIM + col] : 0.0f;
        dst[idx] = to_bf16(v);            // dst[col*RP + row]
    }
}

__global__ void convert_rm(const float* __restrict__ src, __bf16* __restrict__ dst, int n) {
    for (int idx = blockIdx.x * blockDim.x + threadIdx.x; idx < n;
         idx += gridDim.x * blockDim.x)
        dst[idx] = to_bf16(src[idx]);
}

// zero accumulators + zero the pad rows (10000..10015) of Yt / Y2t
__global__ void init_misc(__bf16* __restrict__ Yt, __bf16* __restrict__ Y2t,
                          float* __restrict__ accs) {
    int idx = blockIdx.x * blockDim.x + threadIdx.x;
    if (idx < 4) accs[idx] = 0.0f;
    if (idx < EDIM * 16) {
        int col = idx >> 4, row = NROW + (idx & 15);
        __bf16 z = to_bf16(0.0f);
        Yt [(size_t)col * KP + row] = z;
        Y2t[(size_t)col * KP + row] = z;
    }
}

// ---------------------------------------------------------------------------
// Big GEMM: T[M x 64] = Abf[M x KP] @ Xt (Xt stored col-major, pitch KP).
// 8 waves/block, wave = 16 rows x 64 cols, bf16 WMMA, f32 accumulate.
// mode 0: out(bf16, col-major KP) = dinv[r] * T                      (Y)
// mode 1: out(bf16, row-major 64) = relu(dinv[r]*(T + Yprev))        (H)
// mode 2: out(f32 , row-major 64) = relu(dinv[r]*(T + Yprev))        (P_G)
// ---------------------------------------------------------------------------
__global__ __launch_bounds__(256)
void gemm_bigA(const __bf16* __restrict__ Abf, const __bf16* __restrict__ Xt,
               const float* __restrict__ dinv, const __bf16* __restrict__ Yprev,
               void* __restrict__ outp, int mode) {
    int lane = threadIdx.x & 31;
    int wave = threadIdx.x >> 5;
    int r0 = blockIdx.x * 128 + wave * 16;
    if (r0 >= NROW) return;

    int m15 = lane & 15;
    int h   = lane >> 4;
    const __bf16* aptr  = Abf + (size_t)(r0 + m15) * KP + 16 * h;  // +8 for hi chunk
    const __bf16* bbase = Xt  + (size_t)m15 * KP + 16 * h;

    v8f acc[4] = {v8f{}, v8f{}, v8f{}, v8f{}};

#pragma unroll 1     // keep accumulators in place: no rotation movs, no hazard nops
    for (int k0 = 0; k0 < KP; k0 += 32) {
        v8bf alo = *(const v8bf*)(aptr + k0);
        v8bf ahi = *(const v8bf*)(aptr + k0 + 8);
        v16bf a;
#pragma unroll
        for (int i = 0; i < 8; i++) { a[i] = alo[i]; a[8 + i] = ahi[i]; }
        __builtin_prefetch(aptr + k0 + 1024, 0, 3);   // WGP-scope prefetch of A stream
#pragma unroll
        for (int ct = 0; ct < 4; ct++) {
            v16bf b = *(const v16bf*)(bbase + (size_t)(ct * 16) * KP + k0);
            acc[ct] = __builtin_amdgcn_wmma_f32_16x16x32_bf16(
                false, a, false, b, (short)0, acc[ct], false, false);
        }
    }

    float dv[8];
#pragma unroll
    for (int v = 0; v < 8; v++) dv[v] = dinv[r0 + 8 * h + v];

#pragma unroll
    for (int ct = 0; ct < 4; ct++) {
        int col = ct * 16 + m15;
        if (mode == 0) {
            v8bf o;
#pragma unroll
            for (int v = 0; v < 8; v++) o[v] = to_bf16(dv[v] * acc[ct][v]);
            *(v8bf*)((__bf16*)outp + (size_t)col * KP + r0 + 8 * h) = o;
        } else if (mode == 1) {
            v8bf yp = *(const v8bf*)(Yprev + (size_t)col * KP + r0 + 8 * h);
            __bf16* H = (__bf16*)outp;
#pragma unroll
            for (int v = 0; v < 8; v++) {
                float t = dv[v] * (acc[ct][v] + from_bf16(yp[v]));
                H[(size_t)(r0 + 8 * h + v) * EDIM + col] = to_bf16(fmaxf(t, 0.0f));
            }
        } else {
            v8bf yp = *(const v8bf*)(Yprev + (size_t)col * KP + r0 + 8 * h);
            float* PG = (float*)outp;
#pragma unroll
            for (int v = 0; v < 8; v++) {
                float t = dv[v] * (acc[ct][v] + from_bf16(yp[v]));
                PG[(size_t)(r0 + 8 * h + v) * EDIM + col] = fmaxf(t, 0.0f);
            }
        }
    }
}

// ---------------------------------------------------------------------------
// Small GEMM: T[M x 64] = Arow[M x 64] @ Wt (Wt col-major 64x64).
// mode 0: out(bf16, col-major KP) = dinv[r] * T                      (Y2)
// mode 1: out(bf16, row-major)    = relu(T + bias)                   (P_F1)
// mode 2: out(f32 , row-major)    = relu(T + bias)                   (P_F)
// ---------------------------------------------------------------------------
__global__ __launch_bounds__(256)
void gemm_small(const __bf16* __restrict__ Arow, const __bf16* __restrict__ Wt,
                const float* __restrict__ dinv, const float* __restrict__ bias,
                void* __restrict__ outp, int mode) {
    int lane = threadIdx.x & 31;
    int wave = threadIdx.x >> 5;
    int r0 = blockIdx.x * 128 + wave * 16;
    if (r0 >= NROW) return;

    int m15 = lane & 15;
    int h   = lane >> 4;
    const __bf16* aptr = Arow + (size_t)(r0 + m15) * EDIM + 16 * h;

    v8f acc[4] = {v8f{}, v8f{}, v8f{}, v8f{}};
#pragma unroll
    for (int k0 = 0; k0 < EDIM; k0 += 32) {
        v8bf alo = *(const v8bf*)(aptr + k0);
        v8bf ahi = *(const v8bf*)(aptr + k0 + 8);
        v16bf a;
#pragma unroll
        for (int i = 0; i < 8; i++) { a[i] = alo[i]; a[8 + i] = ahi[i]; }
#pragma unroll
        for (int ct = 0; ct < 4; ct++) {
            v16bf b = *(const v16bf*)(Wt + (size_t)(ct * 16 + m15) * EDIM + 16 * h + k0);
            acc[ct] = __builtin_amdgcn_wmma_f32_16x16x32_bf16(
                false, a, false, b, (short)0, acc[ct], false, false);
        }
    }

#pragma unroll
    for (int ct = 0; ct < 4; ct++) {
        int col = ct * 16 + m15;
        if (mode == 0) {
            v8bf o;
#pragma unroll
            for (int v = 0; v < 8; v++)
                o[v] = to_bf16(dinv[r0 + 8 * h + v] * acc[ct][v]);
            *(v8bf*)((__bf16*)outp + (size_t)col * KP + r0 + 8 * h) = o;
        } else if (mode == 1) {
            __bf16* O = (__bf16*)outp;
#pragma unroll
            for (int v = 0; v < 8; v++) {
                int row = r0 + 8 * h + v;
                float t = acc[ct][v] + bias[(size_t)row * EDIM + col];
                O[(size_t)row * EDIM + col] = to_bf16(fmaxf(t, 0.0f));
            }
        } else {
            float* O = (float*)outp;
#pragma unroll
            for (int v = 0; v < 8; v++) {
                int row = r0 + 8 * h + v;
                float t = acc[ct][v] + bias[(size_t)row * EDIM + col];
                O[(size_t)row * EDIM + col] = fmaxf(t, 0.0f);
            }
        }
    }
}

// ---------------------------------------------------------------------------
// Reductions
// ---------------------------------------------------------------------------
__global__ __launch_bounds__(256)
void reduce_gf(const float* __restrict__ PG, const float* __restrict__ PF,
               float* __restrict__ P, float* __restrict__ acc) {
    float s = 0.0f;
    const int n = NROW * EDIM;
    for (int i = blockIdx.x * blockDim.x + threadIdx.x; i < n;
         i += gridDim.x * blockDim.x) {
        float g = PG[i], f = PF[i];
        float d = g - f;
        s += d * d;
        P[i] = g + f;
    }
    s = block_reduce_sum(s);
    if (threadIdx.x == 0) atomicAdd(acc, s);
}

__global__ __launch_bounds__(256)
void reduce_pairs(const int* __restrict__ pairs, const float* __restrict__ label,
                  const float* __restrict__ P, float* __restrict__ acc) {
    int wid  = (blockIdx.x * blockDim.x + threadIdx.x) >> 5;
    int lane = threadIdx.x & 31;
    if (wid >= BPAIR) return;
    int i = pairs[wid * 2], j = pairs[wid * 2 + 1];
    const float* Pi = P + (size_t)i * EDIM;
    const float* Pj = P + (size_t)j * EDIM;
    float s = Pi[lane] * Pj[lane] + Pi[lane + 32] * Pj[lane + 32];
    for (int off = 16; off; off >>= 1) s += __shfl_down(s, off, 32);
    if (lane == 0) {
        float d = s - label[wid];
        atomicAdd(acc, 0.5f * d * d);
    }
}

__global__ __launch_bounds__(256)
void reduce_sq(const float* __restrict__ w, int n, float* __restrict__ acc) {
    float s = 0.0f;
    for (int i = blockIdx.x * blockDim.x + threadIdx.x; i < n;
         i += gridDim.x * blockDim.x)
        s += w[i] * w[i];
    s = block_reduce_sum(s);
    if (threadIdx.x == 0) atomicAdd(acc, s);
}

__global__ void finalize_k(const float* __restrict__ accs, float* __restrict__ out) {
    if (threadIdx.x == 0 && blockIdx.x == 0)
        out[0] = LAM2 * accs[0] + accs[1] + LAM1 * accs[2];
}

// ---------------------------------------------------------------------------
extern "C" void kernel_launch(void* const* d_in, const int* in_sizes, int n_in,
                              void* d_out, int out_size, void* d_ws, size_t ws_size,
                              hipStream_t stream) {
    const float* A     = (const float*)d_in[0];
    const int*   pairs = (const int*)  d_in[1];
    const float* label = (const float*)d_in[2];
    const float* W_G1  = (const float*)d_in[3];
    const float* W_G2  = (const float*)d_in[4];
    const float* F     = (const float*)d_in[5];
    const float* W_F1  = (const float*)d_in[6];
    const float* W_F2  = (const float*)d_in[7];
    const float* b1    = (const float*)d_in[8];
    const float* b2    = (const float*)d_in[9];

    char* ws = (char*)d_ws;
    size_t off = 0;
    auto alloc = [&](size_t bytes) -> char* {
        char* p = ws + off;
        off += (bytes + 255) & ~(size_t)255;
        return p;
    };
    __bf16* Abf   = (__bf16*)alloc((size_t)NROW * KP * 2);
    __bf16* WG1t  = (__bf16*)alloc((size_t)EDIM * KP * 2);
    __bf16* Yt    = (__bf16*)alloc((size_t)EDIM * KP * 2);
    __bf16* Y2t   = (__bf16*)alloc((size_t)EDIM * KP * 2);
    __bf16* WG2t  = (__bf16*)alloc((size_t)EDIM * EDIM * 2);
    __bf16* WF1t  = (__bf16*)alloc((size_t)EDIM * EDIM * 2);
    __bf16* WF2t  = (__bf16*)alloc((size_t)EDIM * EDIM * 2);
    __bf16* Hbf   = (__bf16*)alloc((size_t)NROW * EDIM * 2);
    __bf16* Fbf   = (__bf16*)alloc((size_t)NROW * EDIM * 2);
    __bf16* PF1bf = (__bf16*)alloc((size_t)NROW * EDIM * 2);
    float*  PG    = (float*) alloc((size_t)NROW * EDIM * 4);
    float*  PF    = (float*) alloc((size_t)NROW * EDIM * 4);
    float*  P     = (float*) alloc((size_t)NROW * EDIM * 4);
    float*  dinv  = (float*) alloc((size_t)NROW * 4);
    float*  accs  = (float*) alloc(256);

    const int MB = (NROW + 127) / 128;   // 79 blocks for the WMMA GEMMs

    init_misc<<<4, 256, 0, stream>>>(Yt, Y2t, accs);
    rowsum_convert<<<NROW, 256, 0, stream>>>(A, Abf, dinv);

    convert_T<<<512, 256, 0, stream>>>(W_G1, WG1t, NROW, KP);
    convert_T<<<4, 256, 0, stream>>>(W_G2, WG2t, EDIM, EDIM);
    convert_T<<<4, 256, 0, stream>>>(W_F1, WF1t, EDIM, EDIM);
    convert_T<<<4, 256, 0, stream>>>(W_F2, WF2t, EDIM, EDIM);
    convert_rm<<<512, 256, 0, stream>>>(F, Fbf, NROW * EDIM);

    // GCN branch: G1 = A@W_G1 ; Y = dinv*G1
    gemm_bigA<<<MB, 256, 0, stream>>>(Abf, WG1t, dinv, Yt, (void*)Yt, 0);
    // H = relu(dinv*(A@Y + Y))
    gemm_bigA<<<MB, 256, 0, stream>>>(Abf, Yt, dinv, Yt, (void*)Hbf, 1);
    // Y2 = dinv*(H@W_G2)
    gemm_small<<<MB, 256, 0, stream>>>(Hbf, WG2t, dinv, nullptr, (void*)Y2t, 0);
    // P_G = relu(dinv*(A@Y2 + Y2))
    gemm_bigA<<<MB, 256, 0, stream>>>(Abf, Y2t, dinv, Y2t, (void*)PG, 2);

    // MLP branch
    gemm_small<<<MB, 256, 0, stream>>>(Fbf, WF1t, nullptr, b1, (void*)PF1bf, 1);
    gemm_small<<<MB, 256, 0, stream>>>(PF1bf, WF2t, nullptr, b2, (void*)PF, 2);

    // losses
    reduce_gf<<<1024, 256, 0, stream>>>(PG, PF, P, accs + 0);
    reduce_pairs<<<BPAIR / 8, 256, 0, stream>>>(pairs, label, P, accs + 1);
    reduce_sq<<<256, 256, 0, stream>>>(W_G1, NROW * EDIM, accs + 2);
    reduce_sq<<<8, 256, 0, stream>>>(W_G2, EDIM * EDIM, accs + 2);
    reduce_sq<<<8, 256, 0, stream>>>(W_F1, EDIM * EDIM, accs + 2);
    reduce_sq<<<8, 256, 0, stream>>>(W_F2, EDIM * EDIM, accs + 2);

    finalize_k<<<1, 32, 0, stream>>>(accs, (float*)d_out);
}
